// LSTMP_72507637891798
// MI455X (gfx1250) — matrix operand order repaired and checked by reading, not compiled, not run on hardware
//
#include <hip/hip_runtime.h>

typedef unsigned short u16;
typedef __bf16 v16bf __attribute__((ext_vector_type(16)));
typedef float  v8f   __attribute__((ext_vector_type(8)));

#define T_STEPS 512
#define NB      32
#define DIN     512
#define DH      1024
#define DP      512
#define G4      4096   // 4*DH

#define Y_ELEMS   ((size_t)T_STEPS * NB * DP)     // 8388608
#define HT_OFF    (Y_ELEMS)                       // [1,N,PROJ]
#define CT_OFF    (Y_ELEMS + (size_t)NB * DP)     // [1,N,HIDDEN]

// ---------------- workspace layout (bytes) ----------------
#define WS_BAR     0                                   // barrier counter
#define WS_XG      256                                 // f32 [16384,4096]
#define WS_INBF    (WS_XG   + (size_t)16384*4096*4)    // bf16 [16384,512]
#define WS_WIHBF   (WS_INBF + (size_t)16384*512*2)     // bf16 [4096,512]
#define WS_WHHBF   (WS_WIHBF+ (size_t)4096*512*2)      // bf16 [4096,512]
#define WS_WHRBF   (WS_WHHBF+ (size_t)4096*512*2)      // bf16 [512,1024]
#define WS_CST     (WS_WHRBF+ (size_t)512*1024*2)      // f32 [32,1024]
#define WS_HBF     (WS_CST  + (size_t)32*1024*4)       // bf16 [32,512]
#define WS_HPREBF  (WS_HBF  + (size_t)32*512*2)        // bf16 [32,1024]

// padded LDS row strides (ushorts): stride%128 == 8 -> lane bank skew of 4 dwords
#define PH 520    // rows of 512
#define PR 1032   // rows of 1024

__device__ __forceinline__ u16 f2bf(float f) {
    unsigned int u = __float_as_uint(f);
    u += 0x7FFFu + ((u >> 16) & 1u);          // round-to-nearest-even
    return (u16)(u >> 16);
}
__device__ __forceinline__ float sigm(float x) {
    return 1.0f / (1.0f + __expf(-x));
}

union Frag {               // 16 bf16 = 32 bytes = 8 VGPRs
    v16bf v;
    uint4 q[2];
};

// A fragment (16x32, M x K), row-major bf16, leading dim ld (ushorts).
__device__ __forceinline__ Frag load_a(const u16* base, int row0, int ld,
                                       int k0, int lane) {
    Frag f;
    int half = lane >> 4, m = lane & 15;
    const u16* p = base + (size_t)(row0 + m) * ld + k0 + half * 8;
    f.q[0] = *(const uint4*)(p);
    f.q[1] = *(const uint4*)(p + 16);
    return f;
}

// B fragment (32x16, K x N) from weights stored [out_col][K].
__device__ __forceinline__ Frag load_b(const u16* base, int col0, int ld,
                                       int k0, int lane) {
    Frag f;
    int n = lane & 15, kg = lane >> 4;
    const u16* p = base + (size_t)(col0 + n) * ld + k0 + kg * 8;
    f.q[0] = *(const uint4*)(p);
    f.q[1] = *(const uint4*)(p + 16);
    return f;
}

__device__ __forceinline__ v8f wmma_bf16(const Frag& a, const Frag& b, v8f c) {
    return __builtin_amdgcn_wmma_f32_16x16x32_bf16(
        false, a.v, false, b.v, (short)0, c, false, false);
}

// CDNA5 async LDS DMA: per-lane 16B global -> LDS, tracked by ASYNCcnt.
__device__ __forceinline__ void async_g2l_b128(void* lds_ptr, const void* gptr) {
    unsigned lds_off = (unsigned)(size_t)lds_ptr;       // low 32 bits = LDS offset
    asm volatile("global_load_async_to_lds_b128 %0, %1, off"
                 :: "v"(lds_off), "v"(gptr) : "memory");
}
__device__ __forceinline__ void wait_async0() {
    asm volatile("s_wait_asynccnt 0x0" ::: "memory");
}

// Ticket-based grid barrier across nblk co-resident workgroups.
__device__ __forceinline__ void grid_sync(unsigned* cnt, unsigned nblk) {
    __threadfence();                 // make this thread's stores visible (L2)
    __syncthreads();
    if (threadIdx.x == 0) {
        unsigned old = __hip_atomic_fetch_add(cnt, 1u, __ATOMIC_ACQ_REL,
                                              __HIP_MEMORY_SCOPE_AGENT);
        unsigned need = (old / nblk + 1u) * nblk;
        while (__hip_atomic_load(cnt, __ATOMIC_ACQUIRE,
                                 __HIP_MEMORY_SCOPE_AGENT) < need) {
            __builtin_amdgcn_s_sleep(1);
        }
    }
    __syncthreads();
}

// ------------------------------------------------------------------
// Prep: fp32 -> bf16 weight/input copies, state init, barrier reset.
// ------------------------------------------------------------------
__global__ void lstmp_prep(const float* in, const float* h, const float* c,
                           const float* wih, const float* whh, const float* whr,
                           u16* inbf, u16* wihbf, u16* whhbf, u16* whrbf,
                           u16* hbf, float* cst, unsigned* bar) {
    size_t i = (size_t)blockIdx.x * blockDim.x + threadIdx.x;
    size_t stride = (size_t)gridDim.x * blockDim.x;
    if (i == 0) *bar = 0u;
    for (size_t k = i; k < (size_t)16384 * 512; k += stride) inbf[k]  = f2bf(in[k]);
    for (size_t k = i; k < (size_t)4096  * 512; k += stride) wihbf[k] = f2bf(wih[k]);
    for (size_t k = i; k < (size_t)4096  * 512; k += stride) whhbf[k] = f2bf(whh[k]);
    for (size_t k = i; k < (size_t)512  * 1024; k += stride) whrbf[k] = f2bf(whr[k]);
    for (size_t k = i; k < (size_t)32   * 512;  k += stride) hbf[k]   = f2bf(h[k]);
    for (size_t k = i; k < (size_t)32   * 1024; k += stride) cst[k]   = c[k];
}

// ------------------------------------------------------------------
// Batched input GEMM: x_gates[16384,4096] = input[16384,512] @ w_ih^T + b
// ------------------------------------------------------------------
__global__ __launch_bounds__(256)
void lstmp_xgates(const u16* __restrict__ Abf, const u16* __restrict__ Bbf,
                  const float* __restrict__ bias, float* __restrict__ C) {
    const int wave = threadIdx.x >> 5, lane = threadIdx.x & 31;
    const int r0 = blockIdx.x * 128 + (wave & 3) * 32;
    const int c0 = blockIdx.y * 64 + (wave >> 2) * 32;
    v8f acc[2][2];
    for (int mi = 0; mi < 2; ++mi)
        for (int ni = 0; ni < 2; ++ni)
            for (int r = 0; r < 8; ++r) acc[mi][ni][r] = 0.0f;

    for (int k0 = 0; k0 < DIN; k0 += 32) {
        Frag a0 = load_a(Abf, r0,      DIN, k0, lane);
        Frag a1 = load_a(Abf, r0 + 16, DIN, k0, lane);
        Frag b0 = load_b(Bbf, c0,      DIN, k0, lane);
        Frag b1 = load_b(Bbf, c0 + 16, DIN, k0, lane);
        acc[0][0] = wmma_bf16(a0, b0, acc[0][0]);
        acc[0][1] = wmma_bf16(a0, b1, acc[0][1]);
        acc[1][0] = wmma_bf16(a1, b0, acc[1][0]);
        acc[1][1] = wmma_bf16(a1, b1, acc[1][1]);
    }
    const int hi = lane >> 4, col = lane & 15;
    for (int mi = 0; mi < 2; ++mi)
        for (int ni = 0; ni < 2; ++ni) {
            float bv = bias[c0 + ni * 16 + col];
            for (int r = 0; r < 8; ++r) {
                size_t row = (size_t)(r0 + mi * 16 + hi * 8 + r);
                C[row * G4 + c0 + ni * 16 + col] = acc[mi][ni][r] + bv;
            }
        }
}

// ------------------------------------------------------------------
// Persistent sequential scan: 32 blocks x 256 thr.
// Block b owns hidden units j in [32b, 32b+32): its 8 waves compute the
// i/f/g/o gate tiles for those units (wave w -> gate w>>1, j-tile w&1),
// so the LSTM cell is block-local (no grid barrier between gates & cell).
// Recurrent weights live in LDS for the whole scan; h / h_pre staged per
// step via async LDS DMA.  2 grid barriers per step.
// ------------------------------------------------------------------
struct ScanLds {
    u16 whh[128][PH];        // block's w_hh slice: [gate*32+jj][K=512]
    u16 whr[16][PR];         // block's w_hr slice: [local col][K=1024]
    u16 hst[32][PH];         // h_{t-1} (staged per step)
    u16 hpre[32][PR];        // sigmoid(o)*tanh(c) (staged per step)
    union {
        float gates[4][32][32];   // gate pre-activations (gates+cell phase)
        float red[8][32][16];     // projection K-split partials (proj phase)
    } u;
};

__global__ __launch_bounds__(256)
void lstmp_scan(const float* __restrict__ xg,
                const u16* __restrict__ whhbf, const u16* __restrict__ whrbf,
                u16* hbf, u16* hprebf, float* cst,
                float* __restrict__ y, unsigned* bar) {
    __shared__ ScanLds L;
    const int tid  = threadIdx.x;
    const int wave = tid >> 5, lane = tid & 31;
    const int hi = lane >> 4, col = lane & 15;
    const int b = blockIdx.x;
    const int gate = wave >> 1, jt = wave & 1;        // gates-phase tile
    const int gc = gate * DH + b * 32 + jt * 16;      // global gates col base
    const int pc = b * 16;                            // proj col base (0..496)

    // ---- one-time: pin recurrent weight slices in LDS (async DMA) ----
    for (int u = tid; u < 8192; u += 256) {           // w_hh: 128 rows x 64 x16B
        int row = u >> 6, seg = u & 63;
        int gcolw = (row >> 5) * DH + b * 32 + (row & 31);
        async_g2l_b128(&L.whh[row][seg * 8], whhbf + (size_t)gcolw * DP + seg * 8);
    }
    for (int u = tid; u < 2048; u += 256) {           // w_hr: 16 rows x 128 x16B
        int row = u >> 7, seg = u & 127;
        async_g2l_b128(&L.whr[row][seg * 8], whrbf + (size_t)(pc + row) * DH + seg * 8);
    }
    wait_async0();
    __syncthreads();

    for (int t = 0; t < T_STEPS; ++t) {
        const float* xgt = xg + (size_t)t * NB * G4;

        // ---- stage h_{t-1} into LDS (32KB, async DMA) ----
        for (int u = tid; u < 1024; u += 256) {       // 32 rows x 32 x16B
            int row = u >> 5, seg = u & 31;
            async_g2l_b128(&L.hst[row][seg * 8], hbf + (size_t)row * DP + seg * 8);
        }
        wait_async0();
        __syncthreads();

        // ---- gates tile = x_gates[t] + h @ w_hh^T (operands in LDS) ----
        v8f g0, g1;
        for (int r = 0; r < 8; ++r) {
            g0[r] = xgt[(size_t)(hi * 8 + r) * G4 + gc + col];
            g1[r] = xgt[(size_t)(16 + hi * 8 + r) * G4 + gc + col];
        }
        for (int k0 = 0; k0 < DP; k0 += 32) {
            Frag a0 = load_a(&L.hst[0][0], 0,  PH, k0, lane);
            Frag a1 = load_a(&L.hst[0][0], 16, PH, k0, lane);
            Frag bb = load_b(&L.whh[0][0], gate * 32 + jt * 16, PH, k0, lane);
            g0 = wmma_bf16(a0, bb, g0);
            g1 = wmma_bf16(a1, bb, g1);
        }
        if (t + 1 < T_STEPS) {   // prefetch next step's x_gates tile (HBM stream)
            const float* nxt = xg + (size_t)(t + 1) * NB * G4;
            __builtin_prefetch(nxt + (size_t)lane * G4 + gc, 0, 1);
        }
        for (int r = 0; r < 8; ++r) {
            L.u.gates[gate][hi * 8 + r][jt * 16 + col]      = g0[r];
            L.u.gates[gate][16 + hi * 8 + r][jt * 16 + col] = g1[r];
        }
        __syncthreads();          // block-local: cell reads the gate tiles

        // ---- LSTM cell for this block's 32 hidden units ----
        for (int e = tid; e < NB * 32; e += 256) {
            int n = e >> 5, jj = e & 31;
            int j = b * 32 + jj;
            float ig = L.u.gates[0][n][jj];
            float fg = L.u.gates[1][n][jj];
            float gg = L.u.gates[2][n][jj];
            float og = L.u.gates[3][n][jj];
            int ci = n * DH + j;
            float c1 = sigm(fg) * cst[ci] + sigm(ig) * tanhf(gg);
            cst[ci] = c1;
            float hp = sigm(og) * tanhf(c1);
            hprebf[ci] = f2bf(hp);
            if (t == T_STEPS - 1) y[CT_OFF + ci] = c1;        // cT
        }
        grid_sync(bar, 32);

        // ---- stage h_pre into LDS (64KB, async DMA) ----
        for (int u = tid; u < 2048; u += 256) {       // 32 rows x 64 x16B
            int row = u >> 6, seg = u & 63;
            async_g2l_b128(&L.hpre[row][seg * 8], hprebf + (size_t)row * DH + seg * 8);
        }
        wait_async0();
        __syncthreads();          // also: gates union -> red reuse

        // ---- projection: h1 = h_pre @ w_hr^T, K split across 8 waves ----
        v8f p0, p1;
        for (int r = 0; r < 8; ++r) { p0[r] = 0.0f; p1[r] = 0.0f; }
        const int kbase = wave * 128;
        for (int k0 = kbase; k0 < kbase + 128; k0 += 32) {
            Frag a0 = load_a(&L.hpre[0][0], 0,  PR, k0, lane);
            Frag a1 = load_a(&L.hpre[0][0], 16, PR, k0, lane);
            Frag bb = load_b(&L.whr[0][0], 0, PR, k0, lane);
            p0 = wmma_bf16(a0, bb, p0);
            p1 = wmma_bf16(a1, bb, p1);
        }
        for (int r = 0; r < 8; ++r) {
            L.u.red[wave][hi * 8 + r][col]      = p0[r];
            L.u.red[wave][16 + hi * 8 + r][col] = p1[r];
        }
        __syncthreads();
        for (int e = tid; e < NB * 16; e += 256) {    // 512 outputs / block
            int n = e >> 4, cc = e & 15;
            float s = 0.0f;
            for (int w = 0; w < 8; ++w) s += L.u.red[w][n][cc];
            y[(size_t)t * NB * DP + (size_t)n * DP + pc + cc] = s;
            hbf[n * DP + pc + cc] = f2bf(s);
            if (t == T_STEPS - 1) y[HT_OFF + (size_t)n * DP + pc + cc] = s; // hT
        }
        grid_sync(bar, 32);       // h visible + LDS safe for next step
    }
}

extern "C" void kernel_launch(void* const* d_in, const int* in_sizes, int n_in,
                              void* d_out, int out_size, void* d_ws, size_t ws_size,
                              hipStream_t stream) {
    const float* input = (const float*)d_in[0];
    const float* h     = (const float*)d_in[1];
    const float* c     = (const float*)d_in[2];
    const float* w_ih  = (const float*)d_in[3];
    const float* w_hh  = (const float*)d_in[4];
    const float* b     = (const float*)d_in[5];
    const float* w_hr  = (const float*)d_in[6];
    float* out = (float*)d_out;

    char* ws = (char*)d_ws;
    unsigned* bar   = (unsigned*)(ws + WS_BAR);
    float* xgts     = (float*)(ws + WS_XG);
    u16* inbf       = (u16*)(ws + WS_INBF);
    u16* wihbf      = (u16*)(ws + WS_WIHBF);
    u16* whhbf      = (u16*)(ws + WS_WHHBF);
    u16* whrbf      = (u16*)(ws + WS_WHRBF);
    float* cst      = (float*)(ws + WS_CST);
    u16* hbf        = (u16*)(ws + WS_HBF);
    u16* hprebf     = (u16*)(ws + WS_HPREBF);

    lstmp_prep<<<512, 256, 0, stream>>>(input, h, c, w_ih, w_hh, w_hr,
                                        inbf, wihbf, whhbf, whrbf,
                                        hbf, cst, bar);
    lstmp_xgates<<<dim3(128, 64), 256, 0, stream>>>(inbf, wihbf, b, xgts);
    lstmp_scan<<<32, 256, 0, stream>>>(xgts, whhbf, whrbf,
                                       hbf, hprebf, cst, out, bar);
}
